// GRAPH_AUTOENCODER__20375324852258
// MI455X (gfx1250) — matrix operand order repaired and checked by reading, not compiled, not run on hardware
//
#include <hip/hip_runtime.h>
#include <hip/hip_bf16.h>

// ---------------- problem constants (fixed by the reference) ----------------
#define NND   32768      // nodes
#define FD    128        // input feature dim
#define HD    256        // hidden dim
#define NB    256        // graphs
#define MAXN  128        // nodes per graph
#define NE    524288     // edges
#define EPG   2048       // edges per graph  (NE / NB)

typedef _Float16 h8  __attribute__((ext_vector_type(8)));
typedef _Float16 h16 __attribute__((ext_vector_type(16)));
typedef float    f8  __attribute__((ext_vector_type(8)));

// ---------------------------------------------------------------------------
// Generic wave-level WMMA GEMM:  C[M x N] = A[M x K] * Bt[N x K]^T (+bias,act)
// A is f16 row-major (lda = K).  B is stored TRANSPOSED as Bt[N][K] so every
// lane's fragment slice is contiguous in K (vectorized b128 loads).
// One wave (32 lanes) owns one 16x16 C tile; grid = (N/16, M/16, batch).
// epilogue: 0 = none, 1 = +bias, 2 = +bias,ReLU, 3 = sigmoid + zero diagonal.
// ---------------------------------------------------------------------------
__global__ __launch_bounds__(32)
void gemm16(const _Float16* __restrict__ A, const _Float16* __restrict__ Bt,
            const float* __restrict__ bias, float* __restrict__ C,
            _Float16* __restrict__ Ch,
            int M, int N, int K, long sA, long sB, long sC, int ep)
{
    const int lane = threadIdx.x & 31;
    const int hi   = lane >> 4;        // 0: lanes 0-15, 1: lanes 16-31
    const int l15  = lane & 15;
    const int tn = blockIdx.x, tm = blockIdx.y, g = blockIdx.z;

    const _Float16* Ag = A  + (long)g * sA;
    const _Float16* Bg = Bt + (long)g * sB;
    float*          Cg = C  + (long)g * sC;

    const int arow = tm * 16 + l15;    // A row this lane supplies
    const int bcol = tn * 16 + l15;    // B column this lane supplies

    f8 acc = {};
    for (int kb = 0; kb < K; kb += 32) {
        // A 16x32 f16 fragment (ISA 7.12.2): lane<16 -> K {kb..kb+7, kb+16..kb+23}
        //                                    lane>=16 -> shifted by +8
        const _Float16* ap = Ag + (long)arow * K + kb + 8 * hi;
        h8 alo = *(const h8*)(ap);
        h8 ahi = *(const h8*)(ap + 16);
        // B 32x16 f16 fragment: lane<16 -> K {kb..kb+7, kb+16..kb+23}, lane>=16 -> shifted by +8
        const _Float16* bp = Bg + (long)bcol * K + kb + 8 * hi;
        h8 blo = *(const h8*)(bp);
        h8 bhi = *(const h8*)(bp + 16);
        h16 a = __builtin_shufflevector(alo, ahi, 0,1,2,3,4,5,6,7,8,9,10,11,12,13,14,15);
        h16 b = __builtin_shufflevector(blo, bhi, 0,1,2,3,4,5,6,7,8,9,10,11,12,13,14,15);
        acc = __builtin_amdgcn_wmma_f32_16x16x32_f16(false, a, false, b,
                                                     (short)0, acc, false, false);
    }

    // C/D layout: VGPR r -> row (r + 8*hi), column = lane&15
    #pragma unroll
    for (int r = 0; r < 8; ++r) {
        int mloc = tm * 16 + r + 8 * hi;   // row
        int nloc = tn * 16 + l15;          // col
        float v = acc[r];
        if (ep == 1 || ep == 2) v += bias[nloc];
        if (ep == 2) v = fmaxf(v, 0.0f);
        if (ep == 3) {
            v = 1.0f / (1.0f + __expf(-v));
            if (mloc == nloc) v = 0.0f;    // zero diagonal (per-graph M==N==128)
        }
        long off = (long)mloc * N + nloc;
        Cg[off] = v;
        if (Ch) Ch[(long)g * sC + off] = (_Float16)v;
    }
}

// ------------------------- small utility kernels ---------------------------
__global__ void zero32(unsigned* __restrict__ p, long n)
{
    long i = (long)blockIdx.x * blockDim.x + threadIdx.x;
    if (i < n) p[i] = 0u;
}

__global__ void cvt_h(const float* __restrict__ X, _Float16* __restrict__ Y, long n)
{
    long i = (long)blockIdx.x * blockDim.x + threadIdx.x;
    if (i < n) Y[i] = (_Float16)X[i];
}

// W[K x N] f32 (row-major)  ->  Wt[N x K] f16
__global__ void cvt_ht(const float* __restrict__ W, _Float16* __restrict__ Wt,
                       int K, int Ncols)
{
    long i = (long)blockIdx.x * blockDim.x + threadIdx.x;
    if (i < (long)K * Ncols) {
        int k = (int)(i / Ncols), n = (int)(i % Ncols);
        Wt[(long)n * K + k] = (_Float16)W[i];
    }
}

__global__ void deg_count(const int* __restrict__ dstl, int* __restrict__ deg)
{
    int e = blockIdx.x * blockDim.x + threadIdx.x;
    if (e < NE) atomicAdd(&deg[(e / EPG) * MAXN + dstl[e]], 1);
}

__global__ void deg_inv(const int* __restrict__ deg, float* __restrict__ dinv)
{
    int i = blockIdx.x * blockDim.x + threadIdx.x;
    if (i < NND) dinv[i] = rsqrtf((float)(deg[i] + 1));   // +1 self loop
}

// ---------------------------------------------------------------------------
// GCN aggregation, one workgroup per graph. The 128x256 f32 accumulator tile
// (128 KB) lives entirely in LDS (CDNA5: 320 KB/WGP) -> ds_add_f32 atomics.
// hpre is read (dense h = x@W) and overwritten in place with the aggregated
// pre-BN output (+bias); each block only touches its own graph's 128 rows and
// all reads complete (barrier) before writeback.
// ---------------------------------------------------------------------------
__global__ __launch_bounds__(256)
void gcn_agg(float* hpre, const int* __restrict__ srcl, const int* __restrict__ dstl,
             const float* __restrict__ dinv, const float* __restrict__ bias)
{
    extern __shared__ float agg[];                 // MAXN * HD floats = 128 KB
    const int g = blockIdx.x, tid = threadIdx.x;
    const int w = tid >> 5, lane = tid & 31;

    for (int i = tid; i < MAXN * HD; i += 256) agg[i] = 0.0f;
    __syncthreads();

    const float* dv = dinv + g * MAXN;
    const long   hb = (long)g * MAXN * HD;

    for (int e0 = 0; e0 < EPG; e0 += 8) {          // 8 waves, 1 edge each
        int e = g * EPG + e0 + w;
        int s = srcl[e], d = dstl[e];
        float nrm = dv[s] * dv[d];
        const float4* hs = (const float4*)(hpre + hb + (long)s * HD) + lane * 2;
        float4 v0 = hs[0], v1 = hs[1];
        float* ag = agg + d * HD + lane * 8;
        atomicAdd(ag + 0, v0.x * nrm); atomicAdd(ag + 1, v0.y * nrm);
        atomicAdd(ag + 2, v0.z * nrm); atomicAdd(ag + 3, v0.w * nrm);
        atomicAdd(ag + 4, v1.x * nrm); atomicAdd(ag + 5, v1.y * nrm);
        atomicAdd(ag + 6, v1.z * nrm); atomicAdd(ag + 7, v1.w * nrm);
    }
    __syncthreads();

    for (int n = w; n < MAXN; n += 8) {            // self loops: unique (n,f) owner
        float nrm = dv[n] * dv[n];
        const float* hn = hpre + hb + (long)n * HD;
        float* ag = agg + n * HD;
        #pragma unroll
        for (int i = 0; i < 8; ++i) { int f = lane + 32 * i; ag[f] += hn[f] * nrm; }
    }
    __syncthreads();

    for (int i = tid; i < MAXN * HD; i += 256)
        hpre[hb + i] = agg[i] + bias[i & (HD - 1)];
}

// -------- BatchNorm: column sums (coalesced, few atomics), then apply ------
__global__ __launch_bounds__(256)
void colstats(const float* __restrict__ X, float* gsum, float* gsumsq, int rows)
{
    int c = threadIdx.x;                            // HD == 256 columns
    float s = 0.0f, ss = 0.0f;
    for (int r = blockIdx.x; r < rows; r += gridDim.x) {
        float v = X[(long)r * HD + c];
        s += v; ss += v * v;
    }
    atomicAdd(&gsum[c], s);
    atomicAdd(&gsumsq[c], ss);
}

__global__ void bnfin(const float* gsum, const float* gsumsq,
                      float* mean, float* binv, float rows)
{
    int c = threadIdx.x;
    float m = gsum[c] / rows;
    float v = gsumsq[c] / rows - m * m;
    mean[c] = m;
    binv[c] = rsqrtf(v + 1e-5f);
}

// mode 0: y = bn(x)            (fd path; x already relu'd by GEMM epilogue)
// mode 1: y = relu(relu(bn(x)) + res)   (GCN block)
__global__ void bnapply(const float* __restrict__ X, const float* mean,
                        const float* binv, const float* gam, const float* bet,
                        const float* res, float* outF, _Float16* outH, int mode)
{
    long i = (long)blockIdx.x * blockDim.x + threadIdx.x;
    int  c = (int)(i & (HD - 1));
    float v = (X[i] - mean[c]) * binv[c] * gam[c] + bet[c];
    if (mode == 1) { v = fmaxf(v, 0.0f); v += res[i]; v = fmaxf(v, 0.0f); }
    if (outF) outF[i] = v;
    if (outH) outH[i] = (_Float16)v;
}

// row-wise L2 normalize, one wave per row
__global__ __launch_bounds__(256)
void l2norm(const float* __restrict__ X, float* __restrict__ outF,
            _Float16* __restrict__ outH)
{
    int row  = blockIdx.x * 8 + (threadIdx.x >> 5);
    int lane = threadIdx.x & 31;
    const float* x = X + (long)row * HD;
    float vals[8], s = 0.0f;
    #pragma unroll
    for (int i = 0; i < 8; ++i) { vals[i] = x[lane + 32 * i]; s += vals[i] * vals[i]; }
    #pragma unroll
    for (int off = 16; off; off >>= 1) s += __shfl_xor(s, off, 32);
    float sc = 1.0f / fmaxf(sqrtf(s), 1e-12f);
    #pragma unroll
    for (int i = 0; i < 8; ++i) {
        float v = vals[i] * sc;
        outF[(long)row * HD + lane + 32 * i] = v;
        outH[(long)row * HD + lane + 32 * i] = (_Float16)v;
    }
}

__global__ void maxpool(const float* __restrict__ Z, float* outF, _Float16* outH)
{
    int g = blockIdx.x, c = threadIdx.x;
    const float* z = Z + (long)g * MAXN * HD + c;
    float m = -3.4e38f;
    for (int n = 0; n < MAXN; ++n) m = fmaxf(m, z[(long)n * HD]);
    outF[g * HD + c] = m;
    outH[g * HD + c] = (_Float16)m;
}

// ---------------------------------------------------------------------------
extern "C" void kernel_launch(void* const* d_in, const int* in_sizes, int n_in,
                              void* d_out, int out_size, void* d_ws, size_t ws_size,
                              hipStream_t stream)
{
    // ---- inputs (setup_inputs order) ----
    const float* x       = (const float*)d_in[0];
    const int*   srcl    = (const int*)  d_in[1];
    const int*   dstl    = (const int*)  d_in[2];
    const float* gcn_w[3] = {(const float*)d_in[3], (const float*)d_in[7],  (const float*)d_in[11]};
    const float* gcn_b[3] = {(const float*)d_in[4], (const float*)d_in[8],  (const float*)d_in[12]};
    const float* bn_g[3]  = {(const float*)d_in[5], (const float*)d_in[9],  (const float*)d_in[13]};
    const float* bn_b[3]  = {(const float*)d_in[6], (const float*)d_in[10], (const float*)d_in[14]};
    const float* sc_w0 = (const float*)d_in[15];
    const float* sc_b0 = (const float*)d_in[16];
    const float* edge_w= (const float*)d_in[17];
    const float* fd_w0 = (const float*)d_in[18];
    const float* fd_b0 = (const float*)d_in[19];
    const float* fd_g0 = (const float*)d_in[20];
    const float* fd_be0= (const float*)d_in[21];
    const float* fd_w1 = (const float*)d_in[22];
    const float* fd_b1 = (const float*)d_in[23];
    const float* fd_g1 = (const float*)d_in[24];
    const float* fd_be1= (const float*)d_in[25];
    const float* fd_w2 = (const float*)d_in[26];
    const float* fd_b2 = (const float*)d_in[27];
    const float* ph_w0 = (const float*)d_in[28];
    const float* ph_b0 = (const float*)d_in[29];
    const float* ph_w1 = (const float*)d_in[30];
    const float* ph_b1 = (const float*)d_in[31];

    // ---- outputs (concatenated flat, f32) ----
    float* zOut   = (float*)d_out;                  // 32768*256
    float* adjOut = zOut   + (long)NND * HD;        // 256*128*128
    float* xrOut  = adjOut + (long)NB * MAXN * MAXN;// 32768*128
    float* zgOut  = xrOut  + (long)NND * FD;        // 256*256
    float* zmOut  = zgOut  + (long)NB * HD;         // 256*256

    // ---- workspace carve-up ----
    char* ws = (char*)d_ws;
    size_t off = 0;
    auto alloc = [&](size_t bytes) { char* p = ws + off; off += (bytes + 255) & ~(size_t)255; return p; };
    _Float16* w0t = (_Float16*)alloc((size_t)FD * HD * 2);
    _Float16* sct = (_Float16*)alloc((size_t)FD * HD * 2);
    _Float16* w1t = (_Float16*)alloc((size_t)HD * HD * 2);
    _Float16* w2t = (_Float16*)alloc((size_t)HD * HD * 2);
    _Float16* ewt = (_Float16*)alloc((size_t)HD * HD * 2);
    _Float16* f0t = (_Float16*)alloc((size_t)HD * HD * 2);
    _Float16* f1t = (_Float16*)alloc((size_t)HD * HD * 2);
    _Float16* f2t = (_Float16*)alloc((size_t)HD * FD * 2);
    _Float16* p0t = (_Float16*)alloc((size_t)HD * HD * 2);
    _Float16* p1t = (_Float16*)alloc((size_t)HD * HD * 2);
    _Float16* zh  = (_Float16*)alloc((size_t)NND * HD * 2);   // f16 of current A operand
    _Float16* th  = (_Float16*)alloc((size_t)NND * HD * 2);   // f16 of t = z@edge_w
    _Float16* fdh = (_Float16*)alloc((size_t)NND * HD * 2);   // f16 fd intermediates
    float* gemmC  = (float*)alloc((size_t)NND * HD * 4);      // GEMM out / pre-BN (in-place agg)
    float* res0   = (float*)alloc((size_t)NND * HD * 4);      // block-0 shortcut
    float* zcur   = (float*)alloc((size_t)NND * HD * 4);      // activation between blocks
    int*   deg    = (int*)  alloc((size_t)NND * 4);
    float* dinv   = (float*)alloc((size_t)NND * 4);
    float* gsum   = (float*)alloc((size_t)512 * 4);           // gsum[256] | gsumsq[256]
    float* gsumsq = gsum + 256;
    float* mean   = (float*)alloc((size_t)256 * 4);
    float* binv   = (float*)alloc((size_t)256 * 4);
    _Float16* zgh = (_Float16*)alloc((size_t)NB * HD * 2);
    float* phC    = (float*)alloc((size_t)NB * HD * 4);
    _Float16* p0h = (_Float16*)alloc((size_t)NB * HD * 2);

    const float rows = (float)NND;
    const size_t aggLds = (size_t)MAXN * HD * 4;   // 128 KB dynamic LDS

    // ---- phase 0: precision conversion / weight transposes ----
    cvt_h<<<(NND * FD + 255) / 256, 256, 0, stream>>>(x, zh, (long)NND * FD);
    cvt_ht<<<(FD * HD + 255) / 256, 256, 0, stream>>>(gcn_w[0], w0t, FD, HD);
    cvt_ht<<<(FD * HD + 255) / 256, 256, 0, stream>>>(sc_w0,    sct, FD, HD);
    cvt_ht<<<(HD * HD + 255) / 256, 256, 0, stream>>>(gcn_w[1], w1t, HD, HD);
    cvt_ht<<<(HD * HD + 255) / 256, 256, 0, stream>>>(gcn_w[2], w2t, HD, HD);
    cvt_ht<<<(HD * HD + 255) / 256, 256, 0, stream>>>(edge_w,   ewt, HD, HD);
    cvt_ht<<<(HD * HD + 255) / 256, 256, 0, stream>>>(fd_w0,    f0t, HD, HD);
    cvt_ht<<<(HD * HD + 255) / 256, 256, 0, stream>>>(fd_w1,    f1t, HD, HD);
    cvt_ht<<<(HD * FD + 255) / 256, 256, 0, stream>>>(fd_w2,    f2t, HD, FD);
    cvt_ht<<<(HD * HD + 255) / 256, 256, 0, stream>>>(ph_w0,    p0t, HD, HD);
    cvt_ht<<<(HD * HD + 255) / 256, 256, 0, stream>>>(ph_w1,    p1t, HD, HD);

    // ---- degrees (graph is identical for all layers) ----
    zero32<<<(NND + 255) / 256, 256, 0, stream>>>((unsigned*)deg, NND);
    deg_count<<<NE / 256, 256, 0, stream>>>(dstl, deg);
    deg_inv<<<NND / 256, 256, 0, stream>>>(deg, dinv);

    // ---- GCN block 0 (shortcut = x@sc_w0 + sc_b0) ----
    gemm16<<<dim3(HD / 16, NND / 16, 1), 32, 0, stream>>>(zh, w0t, nullptr, gemmC, nullptr,
                                                          NND, HD, FD, 0, 0, 0, 0);
    gemm16<<<dim3(HD / 16, NND / 16, 1), 32, 0, stream>>>(zh, sct, sc_b0, res0, nullptr,
                                                          NND, HD, FD, 0, 0, 0, 1);
    gcn_agg<<<NB, 256, aggLds, stream>>>(gemmC, srcl, dstl, dinv, gcn_b[0]);
    zero32<<<2, 256, 0, stream>>>((unsigned*)gsum, 512);
    colstats<<<256, 256, 0, stream>>>(gemmC, gsum, gsumsq, NND);
    bnfin<<<1, 256, 0, stream>>>(gsum, gsumsq, mean, binv, rows);
    bnapply<<<NND * HD / 256, 256, 0, stream>>>(gemmC, mean, binv, bn_g[0], bn_b[0],
                                                res0, zcur, zh, 1);

    // ---- GCN blocks 1 & 2 (identity shortcut) ----
    for (int blk = 1; blk <= 2; ++blk) {
        const _Float16* wt = (blk == 1) ? w1t : w2t;
        gemm16<<<dim3(HD / 16, NND / 16, 1), 32, 0, stream>>>(zh, wt, nullptr, gemmC, nullptr,
                                                              NND, HD, HD, 0, 0, 0, 0);
        gcn_agg<<<NB, 256, aggLds, stream>>>(gemmC, srcl, dstl, dinv, gcn_b[blk]);
        zero32<<<2, 256, 0, stream>>>((unsigned*)gsum, 512);
        colstats<<<256, 256, 0, stream>>>(gemmC, gsum, gsumsq, NND);
        bnfin<<<1, 256, 0, stream>>>(gsum, gsumsq, mean, binv, rows);
        bnapply<<<NND * HD / 256, 256, 0, stream>>>(gemmC, mean, binv, bn_g[blk], bn_b[blk],
                                                    zcur, zcur, zh, 1);
    }

    // ---- z = L2-normalize(rows) -> output + f16 copy ----
    l2norm<<<NND / 8, 256, 0, stream>>>(zcur, zOut, zh);

    // ---- adjacency decoder: t = z@edge_w ; adj_g = sigmoid(t_g @ z_g^T), zero diag
    gemm16<<<dim3(HD / 16, NND / 16, 1), 32, 0, stream>>>(zh, ewt, nullptr, gemmC, th,
                                                          NND, HD, HD, 0, 0, 0, 0);
    gemm16<<<dim3(MAXN / 16, MAXN / 16, NB), 32, 0, stream>>>(th, zh, nullptr, adjOut, nullptr,
                                                              MAXN, MAXN, HD,
                                                              (long)MAXN * HD, (long)MAXN * HD,
                                                              (long)MAXN * MAXN, 3);

    // ---- feature decoder: [Linear,ReLU,BN] x2 then Linear ----
    gemm16<<<dim3(HD / 16, NND / 16, 1), 32, 0, stream>>>(zh, f0t, fd_b0, gemmC, nullptr,
                                                          NND, HD, HD, 0, 0, 0, 2);
    zero32<<<2, 256, 0, stream>>>((unsigned*)gsum, 512);
    colstats<<<256, 256, 0, stream>>>(gemmC, gsum, gsumsq, NND);
    bnfin<<<1, 256, 0, stream>>>(gsum, gsumsq, mean, binv, rows);
    bnapply<<<NND * HD / 256, 256, 0, stream>>>(gemmC, mean, binv, fd_g0, fd_be0,
                                                nullptr, nullptr, fdh, 0);

    gemm16<<<dim3(HD / 16, NND / 16, 1), 32, 0, stream>>>(fdh, f1t, fd_b1, gemmC, nullptr,
                                                          NND, HD, HD, 0, 0, 0, 2);
    zero32<<<2, 256, 0, stream>>>((unsigned*)gsum, 512);
    colstats<<<256, 256, 0, stream>>>(gemmC, gsum, gsumsq, NND);
    bnfin<<<1, 256, 0, stream>>>(gsum, gsumsq, mean, binv, rows);
    bnapply<<<NND * HD / 256, 256, 0, stream>>>(gemmC, mean, binv, fd_g1, fd_be1,
                                                nullptr, nullptr, fdh, 0);

    gemm16<<<dim3(FD / 16, NND / 16, 1), 32, 0, stream>>>(fdh, f2t, fd_b2, xrOut, nullptr,
                                                          NND, FD, HD, 0, 0, 0, 1);

    // ---- global max pool + pooled-head MLP ----
    maxpool<<<NB, HD, 0, stream>>>(zOut, zgOut, zgh);
    gemm16<<<dim3(HD / 16, NB / 16, 1), 32, 0, stream>>>(zgh, p0t, ph_b0, phC, p0h,
                                                         NB, HD, HD, 0, 0, 0, 2);
    gemm16<<<dim3(HD / 16, NB / 16, 1), 32, 0, stream>>>(p0h, p1t, ph_b1, zmOut, nullptr,
                                                         NB, HD, HD, 0, 0, 0, 1);
}
